// AttentionLayer_21311627722773
// MI455X (gfx1250) — compile-verified, matrix operand order checked by reading, not run on hardware
//
#include <hip/hip_runtime.h>
#include <hip/hip_bf16.h>

typedef __attribute__((ext_vector_type(16))) _Float16 v16h;
typedef __attribute__((ext_vector_type(8)))  _Float16 v8h;
typedef __attribute__((ext_vector_type(8)))  float    v8f;

union U16 { v16h v; v8h h[2]; _Float16 f[16]; };

__device__ __forceinline__ v8f wmma16(v16h a, v16h b, v8f c) {
  return __builtin_amdgcn_wmma_f32_16x16x32_f16(false, a, false, b, (short)0, c, false, false);
}

// 8 contiguous f32 -> 8 f16 halves
__device__ __forceinline__ void cvt8(_Float16* d, const float* s) {
  float4 a = *(const float4*)(s);
  float4 b = *(const float4*)(s + 4);
  d[0]=(_Float16)a.x; d[1]=(_Float16)a.y; d[2]=(_Float16)a.z; d[3]=(_Float16)a.w;
  d[4]=(_Float16)b.x; d[5]=(_Float16)b.y; d[6]=(_Float16)b.z; d[7]=(_Float16)b.w;
}

// halves 0..7 = p[0..7], halves 8..15 = p[16..23]  (CDNA5 16-bit A/B K-split)
__device__ __forceinline__ v16h load_pair16(const _Float16* p) {
  U16 t;
  t.h[0] = *(const v8h*)(p);
  t.h[1] = *(const v8h*)(p + 16);
  return t.v;
}

// in: [K][Nn] f32 row-major  ->  out: [Nn][K] f16 (transposed)
__global__ void cvt_transpose(const float* __restrict__ in, _Float16* __restrict__ out,
                              int K, int Nn) {
  int i = blockIdx.x * 256 + threadIdx.x;
  if (i < K * Nn) {
    int k = i / Nn, n = i % Nn;
    out[(size_t)n * K + k] = (_Float16)in[i];
  }
}

// ---------------- Edge + attention + scatter kernel ----------------
// 8 waves/block; each wave owns TWO 16-row M-tiles (32 edges) so every
// B fragment load feeds two v_wmma (double arithmetic intensity per wait).
__global__ __launch_bounds__(256) void edge_kernel(
    const float* __restrict__ x, const int* __restrict__ ei,
    const float* __restrict__ e, const float* __restrict__ u,
    const int* __restrict__ batch,
    const _Float16* __restrict__ W1t, const float* __restrict__ b1,
    const _Float16* __restrict__ W2t, const float* __restrict__ b2,
    const _Float16* __restrict__ Wa1t, const float* __restrict__ ba1,
    const _Float16* __restrict__ Wa2t, const float* __restrict__ ba2,
    float* __restrict__ enew_out, float* __restrict__ agg,
    float* __restrict__ edge_agg, int E)
{
  __shared__ __align__(16) _Float16 sHid[8][32 * 128];
  __shared__ __align__(16) _Float16 sEn[8][32 * 64];
  __shared__ int   sDst[8][32];
  __shared__ int   sG[8][32];
  __shared__ float sEagg[16 * 64];

  const int tid  = threadIdx.x;
  const int wave = tid >> 5;
  const int lane = tid & 31;
  const int r    = lane & 15;
  const int hi   = lane >> 4;
  const int koff = hi << 3;

  for (int i = tid; i < 16 * 64; i += 256) sEagg[i] = 0.f;
  __syncthreads();

  const int eb = (blockIdx.x * 8 + wave) * 32;   // 32 edges per wave
  if (eb < E) {
    const int e0 = eb + r, e1 = eb + 16 + r;
    const int c0 = e0 < E ? e0 : E - 1;
    const int c1 = e1 < E ? e1 : E - 1;
    const int s0 = ei[c0], d0 = ei[E + c0];
    const int s1 = ei[c1], d1 = ei[E + c1];
    const int g0 = batch[s0], g1 = batch[s1];
    if (lane < 16) {
      sDst[wave][r] = d0; sDst[wave][16 + r] = d1;
      sG[wave][r]   = g0; sG[wave][16 + r]   = g1;
    }

    const float* segs0[4] = { x + (size_t)s0 * 64, x + (size_t)d0 * 64,
                              e + (size_t)c0 * 64, u + (size_t)g0 * 64 };
    const float* segs1[4] = { x + (size_t)s1 * 64, x + (size_t)d1 * 64,
                              e + (size_t)c1 * 64, u + (size_t)g1 * 64 };

    _Float16* hid = &sHid[wave][0];
    _Float16* en  = &sEn[wave][0];
    const v8f vzero = {};

    // ===== edge MLP layer 1: [32,256] x [256,128], relu =====
    v8f acc0[8], acc1[8];
    #pragma unroll
    for (int nt = 0; nt < 8; ++nt) { acc0[nt] = vzero; acc1[nt] = vzero; }
    #pragma unroll
    for (int kc = 0; kc < 8; ++kc) {
      const int o = ((kc & 1) << 5) + koff;
      U16 a0, a1;
      cvt8(&a0.f[0], segs0[kc >> 1] + o); cvt8(&a0.f[8], segs0[kc >> 1] + o + 16);
      cvt8(&a1.f[0], segs1[kc >> 1] + o); cvt8(&a1.f[8], segs1[kc >> 1] + o + 16);
      #pragma unroll
      for (int nt = 0; nt < 8; ++nt) {
        v16h B = load_pair16(W1t + (size_t)(nt * 16 + r) * 256 + kc * 32 + koff);
        acc0[nt] = wmma16(a0.v, B, acc0[nt]);
        acc1[nt] = wmma16(a1.v, B, acc1[nt]);
      }
    }
    #pragma unroll
    for (int nt = 0; nt < 8; ++nt) {
      const int n = nt * 16 + r;
      const float bb = b1[n];
      #pragma unroll
      for (int v = 0; v < 8; ++v) {
        float x0 = acc0[nt][v] + bb, x1 = acc1[nt][v] + bb;
        hid[(hi * 8 + v) * 128 + n]        = (_Float16)(x0 > 0.f ? x0 : 0.f);
        hid[(16 + hi * 8 + v) * 128 + n]   = (_Float16)(x1 > 0.f ? x1 : 0.f);
      }
    }

    // ===== edge MLP layer 2: [32,128] x [128,64] =====
    v8f acc20[4], acc21[4];
    #pragma unroll
    for (int nt = 0; nt < 4; ++nt) { acc20[nt] = vzero; acc21[nt] = vzero; }
    #pragma unroll
    for (int kc = 0; kc < 4; ++kc) {
      v16h A0 = load_pair16(hid + r * 128 + kc * 32 + koff);
      v16h A1 = load_pair16(hid + (16 + r) * 128 + kc * 32 + koff);
      #pragma unroll
      for (int nt = 0; nt < 4; ++nt) {
        v16h B = load_pair16(W2t + (size_t)(nt * 16 + r) * 128 + kc * 32 + koff);
        acc20[nt] = wmma16(A0, B, acc20[nt]);
        acc21[nt] = wmma16(A1, B, acc21[nt]);
      }
    }
    #pragma unroll
    for (int nt = 0; nt < 4; ++nt) {
      const int n = nt * 16 + r;
      const float bb = b2[n];
      #pragma unroll
      for (int v = 0; v < 8; ++v) {
        const float v0 = acc20[nt][v] + bb, v1 = acc21[nt][v] + bb;
        const int m0 = hi * 8 + v, m1 = 16 + hi * 8 + v;
        en[m0 * 64 + n] = (_Float16)v0;
        en[m1 * 64 + n] = (_Float16)v1;
        if (eb + m0 < E) enew_out[(size_t)(eb + m0) * 64 + n] = v0;
        if (eb + m1 < E) enew_out[(size_t)(eb + m1) * 64 + n] = v1;
      }
    }

    // ===== attention layer 1: concat's e-slot (kc 4,5) replaced by e_new =====
    #pragma unroll
    for (int nt = 0; nt < 8; ++nt) { acc0[nt] = vzero; acc1[nt] = vzero; }
    #pragma unroll
    for (int kc = 0; kc < 8; ++kc) {
      const int o = ((kc & 1) << 5) + koff;
      U16 a0, a1;
      if (kc == 4 || kc == 5) {
        const int eo = ((kc - 4) << 5) + koff;
        a0.v = load_pair16(en + r * 64 + eo);
        a1.v = load_pair16(en + (16 + r) * 64 + eo);
      } else {
        cvt8(&a0.f[0], segs0[kc >> 1] + o); cvt8(&a0.f[8], segs0[kc >> 1] + o + 16);
        cvt8(&a1.f[0], segs1[kc >> 1] + o); cvt8(&a1.f[8], segs1[kc >> 1] + o + 16);
      }
      #pragma unroll
      for (int nt = 0; nt < 8; ++nt) {
        v16h B = load_pair16(Wa1t + (size_t)(nt * 16 + r) * 256 + kc * 32 + koff);
        acc0[nt] = wmma16(a0.v, B, acc0[nt]);
        acc1[nt] = wmma16(a1.v, B, acc1[nt]);
      }
    }
    #pragma unroll
    for (int nt = 0; nt < 8; ++nt) {
      const int n = nt * 16 + r;
      const float bb = ba1[n];
      #pragma unroll
      for (int v = 0; v < 8; ++v) {
        float x0 = acc0[nt][v] + bb, x1 = acc1[nt][v] + bb;
        hid[(hi * 8 + v) * 128 + n]        = (_Float16)(x0 > 0.f ? x0 : 0.f);
        hid[(16 + hi * 8 + v) * 128 + n]   = (_Float16)(x1 > 0.f ? x1 : 0.f);
      }
    }

    // ===== attention layer 2 + sigmoid + scatter =====
    #pragma unroll
    for (int nt = 0; nt < 4; ++nt) { acc20[nt] = vzero; acc21[nt] = vzero; }
    #pragma unroll
    for (int kc = 0; kc < 4; ++kc) {
      v16h A0 = load_pair16(hid + r * 128 + kc * 32 + koff);
      v16h A1 = load_pair16(hid + (16 + r) * 128 + kc * 32 + koff);
      #pragma unroll
      for (int nt = 0; nt < 4; ++nt) {
        v16h B = load_pair16(Wa2t + (size_t)(nt * 16 + r) * 128 + kc * 32 + koff);
        acc20[nt] = wmma16(A0, B, acc20[nt]);
        acc21[nt] = wmma16(A1, B, acc21[nt]);
      }
    }
    #pragma unroll
    for (int nt = 0; nt < 4; ++nt) {
      const int n = nt * 16 + r;
      const float bb = ba2[n];
      #pragma unroll
      for (int v = 0; v < 8; ++v) {
        const int m0 = hi * 8 + v, m1 = 16 + hi * 8 + v;
        if (eb + m0 < E) {
          const float aw = 1.f / (1.f + __expf(-(acc20[nt][v] + bb)));
          const float ev = (float)en[m0 * 64 + n];
          atomicAdd(&agg[(size_t)sDst[wave][m0] * 64 + n], ev * aw);
          atomicAdd(&sEagg[sG[wave][m0] * 64 + n], ev);
        }
        if (eb + m1 < E) {
          const float aw = 1.f / (1.f + __expf(-(acc21[nt][v] + bb)));
          const float ev = (float)en[m1 * 64 + n];
          atomicAdd(&agg[(size_t)sDst[wave][m1] * 64 + n], ev * aw);
          atomicAdd(&sEagg[sG[wave][m1] * 64 + n], ev);
        }
      }
    }
  }
  __syncthreads();
  for (int i = tid; i < 16 * 64; i += 256)
    atomicAdd(&edge_agg[i], sEagg[i]);
}

// ---------------- Node MLP kernel ----------------
__global__ __launch_bounds__(256) void node_kernel(
    const float* __restrict__ x, const float* __restrict__ agg,
    const float* __restrict__ u, const int* __restrict__ batch,
    const _Float16* __restrict__ W1t, const float* __restrict__ b1,
    const _Float16* __restrict__ W2t, const float* __restrict__ b2,
    float* __restrict__ xnew_out, float* __restrict__ node_agg, int N)
{
  __shared__ __align__(16) _Float16 sHid[8][16 * 128];
  __shared__ int   sG[8][16];
  __shared__ float sNagg[16 * 64];

  const int tid  = threadIdx.x;
  const int wave = tid >> 5;
  const int lane = tid & 31;
  const int r    = lane & 15;
  const int hi   = lane >> 4;
  const int koff = hi << 3;

  for (int i = tid; i < 16 * 64; i += 256) sNagg[i] = 0.f;
  __syncthreads();

  const int tile = blockIdx.x * 8 + wave;
  if (tile * 16 < N) {
    const int idx = tile * 16 + r;
    const int c   = idx < N ? idx : N - 1;
    const int g   = batch[c];
    if (lane < 16) sG[wave][r] = g;

    const float* segs[3] = { x + (size_t)c * 64, agg + (size_t)c * 64, u + (size_t)g * 64 };

    v16h A[6];
    #pragma unroll
    for (int kc = 0; kc < 6; ++kc) {
      const float* sp = segs[kc >> 1];
      const int o = ((kc & 1) << 5) + koff;
      U16 t;
      cvt8(&t.f[0], sp + o);
      cvt8(&t.f[8], sp + o + 16);
      A[kc] = t.v;
    }

    _Float16* hid = &sHid[wave][0];
    #pragma unroll
    for (int nt = 0; nt < 8; ++nt) {
      v8f acc = {};
      #pragma unroll
      for (int kc = 0; kc < 6; ++kc)
        acc = wmma16(A[kc], load_pair16(W1t + (size_t)(nt * 16 + r) * 192 + kc * 32 + koff), acc);
      const int n = nt * 16 + r;
      const float bb = b1[n];
      #pragma unroll
      for (int v = 0; v < 8; ++v) {
        float val = acc[v] + bb;
        hid[(hi * 8 + v) * 128 + n] = (_Float16)(val > 0.f ? val : 0.f);
      }
    }

    v16h Ah[4];
    #pragma unroll
    for (int kc = 0; kc < 4; ++kc)
      Ah[kc] = load_pair16(hid + r * 128 + kc * 32 + koff);
    #pragma unroll
    for (int nt = 0; nt < 4; ++nt) {
      v8f acc = {};
      #pragma unroll
      for (int kc = 0; kc < 4; ++kc)
        acc = wmma16(Ah[kc], load_pair16(W2t + (size_t)(nt * 16 + r) * 128 + kc * 32 + koff), acc);
      const int n = nt * 16 + r;
      const float bb = b2[n];
      #pragma unroll
      for (int v = 0; v < 8; ++v) {
        const int m = hi * 8 + v;
        if (tile * 16 + m < N) {
          const float val = acc[v] + bb;
          xnew_out[(size_t)(tile * 16 + m) * 64 + n] = val;
          atomicAdd(&sNagg[sG[wave][m] * 64 + n], val);
        }
      }
    }
  }
  __syncthreads();
  for (int i = tid; i < 16 * 64; i += 256)
    atomicAdd(&node_agg[i], sNagg[i]);
}

// ---------------- Global (graph-level) MLP: G=16 rows = one WMMA tile ----------------
__global__ __launch_bounds__(32) void global_kernel(
    const float* __restrict__ u, const float* __restrict__ node_agg,
    const float* __restrict__ edge_agg,
    const _Float16* __restrict__ W1t, const float* __restrict__ b1,
    const _Float16* __restrict__ W2t, const float* __restrict__ b2,
    float* __restrict__ unew_out, int G)
{
  __shared__ __align__(16) _Float16 sHid[16 * 128];
  const int lane = threadIdx.x & 31;
  const int r    = lane & 15;
  const int hi   = lane >> 4;
  const int koff = hi << 3;

  const int c = r < G ? r : G - 1;
  const float* segs[3] = { u + (size_t)c * 64, node_agg + (size_t)c * 64,
                           edge_agg + (size_t)c * 64 };

  v16h A[6];
  #pragma unroll
  for (int kc = 0; kc < 6; ++kc) {
    const float* sp = segs[kc >> 1];
    const int o = ((kc & 1) << 5) + koff;
    U16 t;
    cvt8(&t.f[0], sp + o);
    cvt8(&t.f[8], sp + o + 16);
    A[kc] = t.v;
  }

  #pragma unroll
  for (int nt = 0; nt < 8; ++nt) {
    v8f acc = {};
    #pragma unroll
    for (int kc = 0; kc < 6; ++kc)
      acc = wmma16(A[kc], load_pair16(W1t + (size_t)(nt * 16 + r) * 192 + kc * 32 + koff), acc);
    const int n = nt * 16 + r;
    const float bb = b1[n];
    #pragma unroll
    for (int v = 0; v < 8; ++v) {
      float val = acc[v] + bb;
      sHid[(hi * 8 + v) * 128 + n] = (_Float16)(val > 0.f ? val : 0.f);
    }
  }

  v16h Ah[4];
  #pragma unroll
  for (int kc = 0; kc < 4; ++kc)
    Ah[kc] = load_pair16(&sHid[r * 128 + kc * 32 + koff]);
  #pragma unroll
  for (int nt = 0; nt < 4; ++nt) {
    v8f acc = {};
    #pragma unroll
    for (int kc = 0; kc < 4; ++kc)
      acc = wmma16(Ah[kc], load_pair16(W2t + (size_t)(nt * 16 + r) * 128 + kc * 32 + koff), acc);
    const int n = nt * 16 + r;
    const float bb = b2[n];
    #pragma unroll
    for (int v = 0; v < 8; ++v) {
      const int m = hi * 8 + v;
      if (m < G) unew_out[(size_t)m * 64 + n] = acc[v] + bb;
    }
  }
}

extern "C" void kernel_launch(void* const* d_in, const int* in_sizes, int n_in,
                              void* d_out, int out_size, void* d_ws, size_t ws_size,
                              hipStream_t stream)
{
  const float* x    = (const float*)d_in[0];
  const int*   ei   = (const int*)  d_in[1];
  const float* e    = (const float*)d_in[2];
  const float* u    = (const float*)d_in[3];
  const int*   batch= (const int*)  d_in[4];
  const float* We1 = (const float*)d_in[5];  const float* be1 = (const float*)d_in[6];
  const float* We2 = (const float*)d_in[7];  const float* be2 = (const float*)d_in[8];
  const float* Wa1 = (const float*)d_in[9];  const float* ba1 = (const float*)d_in[10];
  const float* Wa2 = (const float*)d_in[11]; const float* ba2 = (const float*)d_in[12];
  const float* Wn1 = (const float*)d_in[13]; const float* bn1 = (const float*)d_in[14];
  const float* Wn2 = (const float*)d_in[15]; const float* bn2 = (const float*)d_in[16];
  const float* Wg1 = (const float*)d_in[17]; const float* bg1 = (const float*)d_in[18];
  const float* Wg2 = (const float*)d_in[19]; const float* bg2 = (const float*)d_in[20];

  const int N = in_sizes[4];
  const int E = in_sizes[1] / 2;
  const int G = in_sizes[3] / 64;

  char* ws = (char*)d_ws;
  _Float16* We1t = (_Float16*)(ws + 0);        // 256x128 -> 64 KiB
  _Float16* Wa1t = (_Float16*)(ws + 65536);    // 64 KiB
  _Float16* We2t = (_Float16*)(ws + 131072);   // 16 KiB
  _Float16* Wa2t = (_Float16*)(ws + 147456);   // 16 KiB
  _Float16* Wn1t = (_Float16*)(ws + 163840);   // 48 KiB
  _Float16* Wn2t = (_Float16*)(ws + 212992);   // 16 KiB
  _Float16* Wg1t = (_Float16*)(ws + 229376);   // 48 KiB
  _Float16* Wg2t = (_Float16*)(ws + 278528);   // 16 KiB
  float* agg      = (float*)(ws + 294912);     // N*64 f32
  float* node_agg = agg + (size_t)N * 64;      // G*64 f32
  float* edge_agg = node_agg + (size_t)G * 64; // G*64 f32

  float* xnew_out = (float*)d_out;
  float* enew_out = xnew_out + (size_t)N * 64;
  float* unew_out = enew_out + (size_t)E * 64;

  // Weight prep: f32 [K][N] -> f16 [N][K]
  cvt_transpose<<<(256 * 128 + 255) / 256, 256, 0, stream>>>(We1, We1t, 256, 128);
  cvt_transpose<<<(256 * 128 + 255) / 256, 256, 0, stream>>>(Wa1, Wa1t, 256, 128);
  cvt_transpose<<<(128 * 64 + 255) / 256, 256, 0, stream>>>(We2, We2t, 128, 64);
  cvt_transpose<<<(128 * 64 + 255) / 256, 256, 0, stream>>>(Wa2, Wa2t, 128, 64);
  cvt_transpose<<<(192 * 128 + 255) / 256, 256, 0, stream>>>(Wn1, Wn1t, 192, 128);
  cvt_transpose<<<(128 * 64 + 255) / 256, 256, 0, stream>>>(Wn2, Wn2t, 128, 64);
  cvt_transpose<<<(192 * 128 + 255) / 256, 256, 0, stream>>>(Wg1, Wg1t, 192, 128);
  cvt_transpose<<<(128 * 64 + 255) / 256, 256, 0, stream>>>(Wg2, Wg2t, 128, 64);

  // Zero accumulators (agg + node_agg + edge_agg are contiguous)
  hipMemsetAsync(agg, 0, (size_t)(N + 2 * G) * 64 * sizeof(float), stream);

  const int eBlocks = (E + 255) / 256;   // 256 edges per block (32 per wave)
  edge_kernel<<<eBlocks, 256, 0, stream>>>(x, ei, e, u, batch,
      We1t, be1, We2t, be2, Wa1t, ba1, Wa2t, ba2,
      enew_out, agg, edge_agg, E);

  const int nBlocks = (N + 127) / 128;
  node_kernel<<<nBlocks, 256, 0, stream>>>(x, agg, u, batch,
      Wn1t, bn1, Wn2t, bn2, xnew_out, node_agg, N);

  global_kernel<<<1, 32, 0, stream>>>(u, node_agg, edge_agg,
      Wg1t, bg1, Wg2t, bg2, unew_out, G);
}
